// NativePeepholeLSTM_73418170957853
// MI455X (gfx1250) — matrix operand order repaired and checked by reading, not compiled, not run on hardware
//
#include <hip/hip_runtime.h>

// ---------------------------------------------------------------------------
// Peephole LSTM for MI455X (gfx1250, wave32, WMMA).
//   Phase 1: pack weights to bf16, transposed (n-major) for B-fragment loads.
//   Phase 2: xhat[g] = X @ U_g + b_g  (big parallel bf16 WMMA GEMM).
//   Phase 3: recurrence. 4 batch-groups x 16 column-slice WGs.
//            Each WG keeps a 224KB bf16 slice of all 7 recurrent matrices
//            LDS-resident for all 1024 steps; h/c exchanged via L2 with a
//            monotonic global barrier per batch group (2 barriers/step).
// ---------------------------------------------------------------------------

typedef __attribute__((ext_vector_type(16))) __bf16 v16bf;
typedef __attribute__((ext_vector_type(8)))  __bf16 v8bf;
typedef __attribute__((ext_vector_type(8)))  float  v8f;

#define HID   512
#define INSZ  256
#define BATCH 64
#define SEQ   1024
#define NROW  (BATCH * SEQ)                       // 65536 rows of X

static constexpr size_t GATESZ = (size_t)NROW * HID;   // 33,554,432 elems
static constexpr size_t HS_SZ  = GATESZ;               // hidden_outputs elems

// ---------------- helpers ----------------

__device__ __forceinline__ float sigmoidf_(float x) {
  return 1.0f / (1.0f + __expf(-x));
}

__device__ __forceinline__ v16bf cat8(v8bf a, v8bf b) {
  return __builtin_shufflevector(a, b, 0,1,2,3,4,5,6,7,8,9,10,11,12,13,14,15);
}

// A-fragment (16x32 bf16, MxK): lane m=l&15, hi=l>>4.
//  elems j=0..7  -> K = k0 + hi*8 + j
//  elems j=8..15 -> K = k0 + 16 + hi*8 + (j-8)
__device__ __forceinline__ v16bf load_afrag(const __bf16* row, int k0, int hi) {
  const __bf16* p = row + k0 + hi * 8;
  return cat8(*(const v8bf*)p, *(const v8bf*)(p + 16));
}

// B-fragment (32x16 bf16, KxN): lane n=l&15, hi=l>>4.
//  elems j=0..7 -> K = k0 + hi*8 + j; j=8..15 -> K = k0 + 16 + hi*8 + (j-8)   (col stored n-major, k contiguous)
__device__ __forceinline__ v16bf load_bfrag(const __bf16* col, int k0, int hi) {
  const __bf16* p = col + k0 + hi * 8;
  return cat8(*(const v8bf*)p, *(const v8bf*)(p + 16));
}

__device__ __forceinline__ v8f wmma_bf16(v16bf a, v16bf b, v8f c) {
  return __builtin_amdgcn_wmma_f32_16x16x32_bf16(false, a, false, b,
                                                 (short)0, c, false, false);
}

__device__ __forceinline__ unsigned short bf16_bits(__bf16 v) {
  union { __bf16 b; unsigned short u; } x; x.b = v; return x.u;
}

// ---------------- kernel 1: pack weights to bf16, transposed ----------------
// wpack[mat][n][k] = M[k][n], mat: 0 Wi,1 Wf,2 Wc,3 Wo,4 Pi,5 Pf,6 Po  (512x512)
// upack[g][n][k]   = U_g[k][n], g: 0 i,1 f,2 c,3 o                      (512x256)

__global__ void pack_weights(const float* __restrict__ Wi, const float* __restrict__ Wf,
                             const float* __restrict__ Wc, const float* __restrict__ Wo,
                             const float* __restrict__ Pi, const float* __restrict__ Pf,
                             const float* __restrict__ Po,
                             const float* __restrict__ Ui, const float* __restrict__ Uf,
                             const float* __restrict__ Uc, const float* __restrict__ Uo,
                             __bf16* __restrict__ wpack, __bf16* __restrict__ upack) {
  int idx = blockIdx.x * blockDim.x + threadIdx.x;
  const int WTOT = 7 * HID * HID;   // 1,835,008
  const int UTOT = 4 * HID * INSZ;  //   524,288
  if (idx < WTOT) {
    int mat = idx >> 18;            // / (512*512)
    int r   = idx & 262143;
    int n   = r >> 9;
    int k   = r & 511;
    const float* src;
    switch (mat) {
      case 0: src = Wi; break; case 1: src = Wf; break;
      case 2: src = Wc; break; case 3: src = Wo; break;
      case 4: src = Pi; break; case 5: src = Pf; break;
      default: src = Po; break;
    }
    wpack[idx] = (__bf16)src[k * HID + n];
  } else if (idx < WTOT + UTOT) {
    int j = idx - WTOT;
    int g = j >> 17;                // / (512*256)
    int r = j & 131071;
    int n = r >> 8;
    int k = r & 255;
    const float* src = (g == 0) ? Ui : (g == 1) ? Uf : (g == 2) ? Uc : Uo;
    upack[j] = (__bf16)src[k * HID + n];
  }
}

// ---------------- kernel 2: input projections xhat[g] = X@U_g + b_g ----------
// grid (4096 m-tiles, 4 gates), 256 threads = 8 waves; wave -> 4 n-tiles.

__global__ __launch_bounds__(256)
void xproj(const float* __restrict__ X, const __bf16* __restrict__ upack,
           const float* __restrict__ bi, const float* __restrict__ bf_,
           const float* __restrict__ bc, const float* __restrict__ bo,
           __bf16* __restrict__ xhat) {
  __shared__ __align__(16) __bf16 x_sh[16][INSZ];  // 8KB bf16 tile of X

  const int mt  = blockIdx.x;
  const int g   = blockIdx.y;
  const int tid = threadIdx.x;

  for (int i = 0; i < 16; ++i) {
    int idx = tid + i * 256;
    int row = idx >> 8, col = idx & 255;
    x_sh[row][col] = (__bf16)X[(size_t)(mt * 16 + row) * INSZ + col];
  }
  __syncthreads();

  const int wave = tid >> 5, lane = tid & 31;
  const int hi = lane >> 4, lm = lane & 15;
  const float* bias = (g == 0) ? bi : (g == 1) ? bf_ : (g == 2) ? bc : bo;
  __bf16* out = xhat + (size_t)g * GATESZ;

  for (int ti = 0; ti < 4; ++ti) {
    int nt = wave + ti * 8;
    int n  = nt * 16 + lm;
    v8f acc = {};
    const __bf16* bcol = upack + (size_t)(g * HID + n) * INSZ;
    const __bf16* arow = &x_sh[lm][0];
#pragma unroll
    for (int k0 = 0; k0 < INSZ; k0 += 32)
      acc = wmma_bf16(load_afrag(arow, k0, hi), load_bfrag(bcol, k0, hi), acc);
    float bv = bias[n];
#pragma unroll
    for (int r = 0; r < 8; ++r) {
      int m = r + hi * 8;
      out[(size_t)(mt * 16 + m) * HID + n] = (__bf16)(acc[r] + bv);
    }
  }
}

// ---------------- kernel 3: recurrence ----------------
// 64 WGs x 512 threads (16 waves). WG = (batch-group bg in 0..3, col-slice cs in 0..15).
// LDS-resident bf16 weight slices: w_sl[mat][32 cols][512 k]  (224KB).

__device__ __forceinline__ void groupbar(unsigned* ctr, unsigned target) {
  __syncthreads();
  if (threadIdx.x == 0) {
    __threadfence();
    atomicAdd(ctr, 1u);
    while (atomicAdd(ctr, 0u) < target) __builtin_amdgcn_s_sleep(2);
    __threadfence();
  }
  __syncthreads();
}

__global__ __launch_bounds__(512)
void lstm_rec(const __bf16* __restrict__ wpack, const __bf16* __restrict__ xhat,
              __bf16* cbuf, __bf16* hbuf, unsigned* counters,
              float* __restrict__ out) {
  __shared__ __align__(16) __bf16 w_sl[7][32][HID];     // 229,376 B
  __shared__ __align__(16) __bf16 h_sh[16][HID];        //  16,384 B (full h, bf16)
  __shared__ __align__(16) __bf16 c_sh[16][HID];        //  16,384 B (full c, bf16)
  __shared__ float accb[6][16][32];                     //  12,288 B (phase-A accums)
  __shared__ float acc_pop[4][16][32];                  //   8,192 B (P_o k-partials)
  __shared__ float c_loc[16][32];                       //   2,048 B (fp32 cell slice)

  const int wgid = blockIdx.x;
  const int cs = wgid & 15;           // column slice
  const int bg = wgid >> 4;           // batch group
  const int n0 = cs * 32;
  const int tid = threadIdx.x;
  const int wave = tid >> 5, lane = tid & 31;
  const int hi = lane >> 4, lm = lane & 15;

  // --- load weight slices (uint4 = 8 bf16 per transfer) ---
  {
    const uint4* src = (const uint4*)wpack;
    uint4* dst = (uint4*)&w_sl[0][0][0];
    for (int v = tid; v < 7 * 32 * HID / 8; v += 512) {
      int mat = v >> 11;              // / (32*512/8)
      int r   = v & 2047;
      int nl  = r >> 6;               // / (512/8)
      int kv  = r & 63;
      dst[v] = src[(size_t)mat * 32768 + (size_t)(n0 + nl) * 64 + kv];
    }
  }
  // --- zero-init h, c ---
  for (int i = tid; i < 16 * HID; i += 512) {
    (&h_sh[0][0])[i] = (__bf16)0.0f;
    (&c_sh[0][0])[i] = (__bf16)0.0f;
  }
  const int m  = tid >> 5;            // elementwise row (== wave)
  const int nl = tid & 31;            // elementwise local col
  const int n  = n0 + nl;
  const int b  = bg * 16 + m;
  c_loc[m][nl] = 0.0f;
  __syncthreads();

  __bf16* cgrp = cbuf + (size_t)bg * 16 * HID;
  __bf16* hgrp = hbuf + (size_t)bg * 16 * HID;
  unsigned* ctr = counters + bg;
  unsigned target = 16;

  for (int t = 0; t < SEQ; ++t) {
    // ---- phase A: h@{Wi,Wf,Wc,Wo}, c@{Pi,Pf} for this 32-col slice ----
    if (wave < 12) {
      int mat = wave >> 1, ct = wave & 1;
      const __bf16* A = (mat < 4) ? &h_sh[lm][0] : &c_sh[lm][0];
      int bn = ct * 16 + lm;
      const __bf16* B = &w_sl[mat][bn][0];
      v8f acc = {};
#pragma unroll
      for (int k0 = 0; k0 < HID; k0 += 32)
        acc = wmma_bf16(load_afrag(A, k0, hi), load_bfrag(B, k0, hi), acc);
#pragma unroll
      for (int r = 0; r < 8; ++r) accb[mat][r + hi * 8][bn] = acc[r];
    }
    __syncthreads();

    // ---- elementwise 1: gates i,f,g -> c_new; publish c_new (bf16) ----
    const size_t xidx = ((size_t)b * SEQ + t) * HID + n;
    {
      float xi = (float)xhat[0 * GATESZ + xidx];
      float xf = (float)xhat[1 * GATESZ + xidx];
      float xc = (float)xhat[2 * GATESZ + xidx];
      float pre_i = accb[0][m][nl] + accb[4][m][nl] + xi;
      float pre_f = accb[1][m][nl] + accb[5][m][nl] + xf;
      float pre_g = accb[2][m][nl] + xc;
      float it = sigmoidf_(pre_i);
      float ft = sigmoidf_(pre_f);
      float gt = tanhf(pre_g);
      float cn = ft * c_loc[m][nl] + it * gt;
      c_loc[m][nl] = cn;
      ((volatile unsigned short*)cgrp)[m * HID + n] = bf16_bits((__bf16)cn);
    }
    groupbar(ctr, target); target += 16;

    // ---- gather full c_new (bf16) from the 16 slices of this group ----
    {
      volatile const unsigned* s = (volatile const unsigned*)cgrp;
      unsigned* d = (unsigned*)&c_sh[0][0];
      for (int i = tid; i < 16 * HID / 2; i += 512) d[i] = s[i];
    }
    __syncthreads();

    // ---- phase B: c_new @ P_o (K split in 4 deterministic partials) ----
    if (wave < 8) {
      int ct = wave >> 2, ks = wave & 3;
      int bn = ct * 16 + lm;
      const __bf16* A = &c_sh[lm][0];
      const __bf16* B = &w_sl[6][bn][0];
      v8f acc = {};
#pragma unroll
      for (int k0 = ks * 128; k0 < ks * 128 + 128; k0 += 32)
        acc = wmma_bf16(load_afrag(A, k0, hi), load_bfrag(B, k0, hi), acc);
#pragma unroll
      for (int r = 0; r < 8; ++r) acc_pop[ks][r + hi * 8][bn] = acc[r];
    }
    __syncthreads();

    // ---- elementwise 2: o gate -> h_new; publish h_new ----
    {
      float xo = (float)xhat[3 * GATESZ + xidx];
      float po = accb[3][m][nl] + acc_pop[0][m][nl] + acc_pop[1][m][nl]
               + acc_pop[2][m][nl] + acc_pop[3][m][nl] + xo;
      float ot = sigmoidf_(po);
      float cn = c_loc[m][nl];
      float hn = ot * tanhf(cn);
      out[xidx] = hn;                                   // hidden_outputs
      ((volatile unsigned short*)hgrp)[m * HID + n] = bf16_bits((__bf16)hn);
      if (t == SEQ - 1)
        out[HS_SZ + (size_t)b * HID + n] = hn;          // h_T
      // prefetch next step's xhat lines (global_prefetch_b8)
      if (t + 1 < SEQ) {
        __builtin_prefetch(&xhat[0 * GATESZ + xidx + HID], 0, 1);
        __builtin_prefetch(&xhat[1 * GATESZ + xidx + HID], 0, 1);
        __builtin_prefetch(&xhat[2 * GATESZ + xidx + HID], 0, 1);
        __builtin_prefetch(&xhat[3 * GATESZ + xidx + HID], 0, 1);
      }
    }
    groupbar(ctr, target); target += 16;

    // ---- gather full h_new (bf16) ----
    {
      volatile const unsigned* s = (volatile const unsigned*)hgrp;
      unsigned* d = (unsigned*)&h_sh[0][0];
      for (int i = tid; i < 16 * HID / 2; i += 512) d[i] = s[i];
    }
    __syncthreads();
  }

  // ---- c_T ----
  out[HS_SZ + (size_t)BATCH * HID + (size_t)b * HID + n] = c_loc[m][nl];
}

// ---------------- host-side launch ----------------

extern "C" void kernel_launch(void* const* d_in, const int* in_sizes, int n_in,
                              void* d_out, int out_size, void* d_ws, size_t ws_size,
                              hipStream_t stream) {
  const float* X  = (const float*)d_in[0];
  const float* Ui = (const float*)d_in[1];
  const float* Wi = (const float*)d_in[2];
  const float* bi = (const float*)d_in[3];
  const float* Pi = (const float*)d_in[4];
  const float* Uf = (const float*)d_in[5];
  const float* Wf = (const float*)d_in[6];
  const float* bf = (const float*)d_in[7];
  const float* Pf = (const float*)d_in[8];
  const float* Uc = (const float*)d_in[9];
  const float* Wc = (const float*)d_in[10];
  const float* bc = (const float*)d_in[11];
  const float* Uo = (const float*)d_in[12];
  const float* Wo = (const float*)d_in[13];
  const float* bo = (const float*)d_in[14];
  const float* Po = (const float*)d_in[15];

  // workspace layout (256B-aligned regions)
  constexpr size_t OFF_CTR   = 0;
  constexpr size_t OFF_WPACK = 256;
  constexpr size_t SZ_WPACK  = (size_t)7 * HID * HID * 2;     // 3,670,016
  constexpr size_t OFF_UPACK = OFF_WPACK + SZ_WPACK;
  constexpr size_t SZ_UPACK  = (size_t)4 * HID * INSZ * 2;    // 1,048,576
  constexpr size_t OFF_CBUF  = OFF_UPACK + SZ_UPACK;
  constexpr size_t SZ_XBUF   = (size_t)4 * 16 * HID * 2;      // 65,536
  constexpr size_t OFF_HBUF  = OFF_CBUF + SZ_XBUF;
  constexpr size_t OFF_XHAT  = OFF_HBUF + SZ_XBUF;            // 4 * 64MB bf16

  char* ws = (char*)d_ws;
  unsigned* counters = (unsigned*)(ws + OFF_CTR);
  __bf16* wpack = (__bf16*)(ws + OFF_WPACK);
  __bf16* upack = (__bf16*)(ws + OFF_UPACK);
  __bf16* cbuf  = (__bf16*)(ws + OFF_CBUF);
  __bf16* hbuf  = (__bf16*)(ws + OFF_HBUF);
  __bf16* xhat  = (__bf16*)(ws + OFF_XHAT);

  hipMemsetAsync(ws + OFF_CTR, 0, 256, stream);

  {
    const int total = 7 * HID * HID + 4 * HID * INSZ;  // 2,359,296
    pack_weights<<<dim3((total + 255) / 256), dim3(256), 0, stream>>>(
        Wi, Wf, Wc, Wo, Pi, Pf, Po, Ui, Uf, Uc, Uo, wpack, upack);
  }

  xproj<<<dim3(NROW / 16, 4), dim3(256), 0, stream>>>(X, upack, bi, bf, bc, bo, xhat);

  lstm_rec<<<dim3(64), dim3(512), 0, stream>>>(wpack, xhat, cbuf, hbuf, counters,
                                               (float*)d_out);
}
